// DynamicSnakeConv_22127671509609
// MI455X (gfx1250) — matrix operand order repaired and checked by reading, not compiled, not run on hardware
//
#include <hip/hip_runtime.h>
#include <hip/hip_bf16.h>
#include <math.h>

typedef __attribute__((ext_vector_type(16))) _Float16 v16h;
typedef __attribute__((ext_vector_type(8)))  float    v8f;

#define Bsz   8
#define Cch   128
#define Hh    128
#define Ww    128
#define HW    (Hh * Ww)
#define Ktap  5
#define OUTC  256
#define NPIX  128          // pixels per workgroup tile
#define SSTR  136          // padded LDS row stride (halfs) to spread banks
#define THREADS 512        // 16 waves (wave32)

__global__ __launch_bounds__(THREADS)
void dsc_fused_kernel(const float* __restrict__ x,
                      const float* __restrict__ w_off,
                      const float* __restrict__ b_off,
                      const float* __restrict__ w_fin,
                      const float* __restrict__ b_fin,
                      float* __restrict__ out)
{
    __shared__ _Float16 S[NPIX * SSTR];      // sampled tile [pix][c], f16
    __shared__ float dxs[Ktap][NPIX];
    __shared__ float dys[Ktap][NPIX];
    __shared__ float msk[Ktap][NPIX];
    __shared__ int   gIdx[NPIX][4];          // per-tap gather plane offsets
    __shared__ float gW[NPIX][4];            // mask-folded bilinear weights

    const int tid  = threadIdx.x;
    const int wave = tid >> 5;               // 0..15 -> 16-row OC strip
    const int lane = tid & 31;
    const int mRow = lane & 15;
    const int hsel = lane >> 4;              // lane half (K/M sub-select)

    const long pixBase = (long)blockIdx.x * NPIX;   // flattened B*H*W pixel
    const int  b   = (int)(pixBase / HW);
    const int  pib = (int)(pixBase % HW);           // pixel-in-image base

    // ---------------- Phase A: 1x1 offset conv (15 dots / pixel) ----------
    if (tid < NPIX) {
        const int p = tid;
        const float* xp = x + (size_t)b * Cch * HW + (pib + p);
        float a[3 * Ktap];
        #pragma unroll
        for (int o = 0; o < 3 * Ktap; ++o) a[o] = b_off[o];
        for (int c = 0; c < Cch; ++c) {
            const float xv = xp[(size_t)c * HW];     // coalesced along W
            #pragma unroll
            for (int o = 0; o < 3 * Ktap; ++o)
                a[o] += xv * w_off[o * Cch + c];     // uniform -> scalar loads
        }
        #pragma unroll
        for (int o = 0; o < Ktap; ++o) {
            dxs[o][p] = tanhf(a[o]);
            dys[o][p] = tanhf(a[o + Ktap]);
            msk[o][p] = 1.0f / (1.0f + expf(-a[o + 2 * Ktap]));
        }
    }
    __syncthreads();

    // f32 accumulators: 8 N-tiles of 16x16 per wave (M = 16 OCs of this wave)
    v8f acc[8];
    #pragma unroll
    for (int nt = 0; nt < 8; ++nt)
        #pragma unroll
        for (int j = 0; j < 8; ++j) acc[nt][j] = 0.0f;

    for (int tap = 0; tap < Ktap; ++tap) {
        // -------- per-pixel bilinear-reflect metadata for this tap --------
        if (tid < NPIX) {
            const int p  = tid;
            const int pi = pib + p;
            const float xc = (float)(pi & (Ww - 1));
            const float yc = (float)(pi >> 7);
            float px = xc + 5.0f * dxs[tap][p];      // algebra-simplified ref
            float py = yc + 5.0f * dys[tap][p];
            const float mW = 2.0f * (float)(Ww - 1);
            const float mH = 2.0f * (float)(Hh - 1);
            float ax = fabsf(px); ax -= mW * floorf(ax / mW);
            if (ax > (float)(Ww - 1)) ax = mW - ax;
            float ay = fabsf(py); ay -= mH * floorf(ay / mH);
            if (ay > (float)(Hh - 1)) ay = mH - ay;
            const float x0f = floorf(ax), y0f = floorf(ay);
            const float wx1 = ax - x0f,  wy1 = ay - y0f;
            int x0 = (int)x0f, y0 = (int)y0f;
            const int x0c = min(max(x0, 0), Ww - 1);
            const int x1c = min(max(x0 + 1, 0), Ww - 1);
            const int y0c = min(max(y0, 0), Hh - 1);
            const int y1c = min(max(y0 + 1, 0), Hh - 1);
            const float mv = msk[tap][p];
            gIdx[p][0] = y0c * Ww + x0c;  gIdx[p][1] = y0c * Ww + x1c;
            gIdx[p][2] = y1c * Ww + x0c;  gIdx[p][3] = y1c * Ww + x1c;
            gW[p][0] = (1.0f - wy1) * (1.0f - wx1) * mv;
            gW[p][1] = (1.0f - wy1) * wx1 * mv;
            gW[p][2] = wy1 * (1.0f - wx1) * mv;
            gW[p][3] = wy1 * wx1 * mv;
        }
        __syncthreads();

        // -------- gather [128 px x 128 ch] sample tile into LDS (f16) -----
        for (int e = tid; e < NPIX * Cch; e += THREADS) {
            const int p = e & (NPIX - 1);
            const int c = e >> 7;
            const float* bp = x + ((size_t)b * Cch + c) * HW;   // L2-resident
            const float s = gW[p][0] * bp[gIdx[p][0]]
                          + gW[p][1] * bp[gIdx[p][1]]
                          + gW[p][2] * bp[gIdx[p][2]]
                          + gW[p][3] * bp[gIdx[p][3]];
            S[p * SSTR + c] = (_Float16)s;
        }
        __syncthreads();

        // -------- A fragments: w_fin tap slice, documented 16-bit A layout
        // VGPR v<4: K = hsel*8 + 2v ; v>=4: K = 16 + hsel*8 + 2(v-4)
        v16h afr[4];
        const float* wf = w_fin + (size_t)(wave * 16 + mRow) * (Cch * Ktap)
                                + tap * Cch;
        #pragma unroll
        for (int kc = 0; kc < 4; ++kc) {
            union { v16h v; _Float16 h[16]; } af;
            #pragma unroll
            for (int v = 0; v < 8; ++v) {
                const int k0 = kc * 32 +
                    (v < 4 ? (hsel * 8 + 2 * v) : (16 + hsel * 8 + 2 * (v - 4)));
                af.h[2 * v]     = (_Float16)wf[k0];
                af.h[2 * v + 1] = (_Float16)wf[k0 + 1];
            }
            afr[kc] = af.v;
        }

        // -------- MMA: B layout (doc): elem s<8: K=hsel*8+s ; s>=8: K=16+hsel*8+(s-8)
        #pragma unroll
        for (int nt = 0; nt < 8; ++nt) {
            const int n = nt * 16 + mRow;          // pixel within tile
            #pragma unroll
            for (int kc = 0; kc < 4; ++kc) {
                union { v16h v; unsigned int w[8]; } bf;
                const unsigned int* sp =
                    (const unsigned int*)&S[n * SSTR + kc * 32 + hsel * 8];
                #pragma unroll
                for (int j = 0; j < 8; ++j) bf.w[j] = sp[j + (j & 4)];  // 2x16B spans, b128s
                acc[nt] = __builtin_amdgcn_wmma_f32_16x16x32_f16(
                    false, afr[kc], false, bf.v, (short)0, acc[nt],
                    false, false);
            }
        }
        __syncthreads();   // before next tap rewrites S / metadata
    }

    // ---------------- epilogue: bias + coalesced f32 stores ---------------
    #pragma unroll
    for (int nt = 0; nt < 8; ++nt) {
        const int n = nt * 16 + mRow;
        float* op = out + (size_t)b * OUTC * HW + (pib + n);
        #pragma unroll
        for (int r = 0; r < 8; ++r) {
            const int m = wave * 16 + hsel * 8 + r;   // documented C/D layout
            op[(size_t)m * HW] = acc[nt][r] + b_fin[m];
        }
    }
}

extern "C" void kernel_launch(void* const* d_in, const int* in_sizes, int n_in,
                              void* d_out, int out_size, void* d_ws, size_t ws_size,
                              hipStream_t stream) {
    (void)in_sizes; (void)n_in; (void)out_size; (void)d_ws; (void)ws_size;
    const float* x     = (const float*)d_in[0];
    const float* w_off = (const float*)d_in[1];
    const float* b_off = (const float*)d_in[2];
    const float* w_fin = (const float*)d_in[3];
    const float* b_fin = (const float*)d_in[4];
    float* out = (float*)d_out;

    const int tiles = (Bsz * HW) / NPIX;   // 1024 workgroups
    hipLaunchKernelGGL(dsc_fused_kernel, dim3(tiles), dim3(THREADS), 0, stream,
                       x, w_off, b_off, w_fin, b_fin, out);
}
